// AttentionBlock_8753143349398
// MI455X (gfx1250) — matrix operand order repaired and checked by reading, not compiled, not run on hardware
//
#include <hip/hip_runtime.h>

typedef __attribute__((ext_vector_type(16))) _Float16 v16h;
typedef __attribute__((ext_vector_type(8)))  float    v8f;

#define WMMA_F16(a, b, c) \
  __builtin_amdgcn_wmma_f32_16x16x32_f16(false, (a), false, (b), (short)0, (c), false, false)

static constexpr int kB = 4;
static constexpr int kC = 256;
static constexpr int kD = 32;
static constexpr int kN = 64 * 64;      // 4096 tokens per batch
static constexpr int kRows = kB * kN;   // 16384 total rows

// ===========================================================================
// Fragment-linear ("packed") layouts. Every fragment is 512 halfs = 1 KB:
//   frag[lane][j], lane = 0..31, j = 0..15  ->  one contiguous 32 B per lane.
//
// A-layout (16x32):  lane&15 = M ; lane<16 -> K in {0..7,16..23} else
//                    {8..15,24..31}; element j -> K = khalf + (j<8 ? j : 8+j)
// B-layout (32x16):  lane&15 = N ; khalf = (lane<16)?0:8 ; j -> K = khalf + (j<8 ? j : 8+j)
// C/D (16x16 f32):   VGPR v -> M = v + (lane<16?0:8), N = lane&15
// ===========================================================================

// ---- pack x (f32, row-major [kRows, kC]) into A-fragment order ------------
__global__ void pack_x_a_kernel(const float* __restrict__ x,
                                _Float16* __restrict__ xA) {
  int idx = blockIdx.x * 256 + threadIdx.x;      // < kRows*kC
  int t  = idx >> 12;                            // 4096 halfs per row tile
  int kk = (idx >> 9) & 7;
  int L  = (idx >> 4) & 31;
  int j  = idx & 15;
  int m  = L & 15;
  int khalf = (L < 16) ? 0 : 8;
  int k = kk * 32 + khalf + ((j < 8) ? j : 8 + j);
  xA[idx] = (_Float16)x[((size_t)t * 16 + m) * kC + k];
}

// ---- pack weight (f32, row-major [kC, O]) into B-fragment order -----------
__global__ void pack_w_b_kernel(const float* __restrict__ w,
                                _Float16* __restrict__ wB, int O) {
  int idx = blockIdx.x * 256 + threadIdx.x;
  if (idx >= kC * O) return;
  int fragIdx = idx >> 9;
  int L = (idx >> 4) & 31;
  int j = idx & 15;
  int NT = O >> 4;
  int kt = fragIdx / NT, nt = fragIdx - kt * NT;
  int k = kt * 32 + ((L < 16) ? 0 : 8) + ((j < 8) ? j : 8 + j);
  int n = nt * 16 + (L & 15);
  wB[idx] = (_Float16)w[(size_t)k * O + n];
}

// ===========================================================================
// QKV projection. One block per 16-row tile (1024), 4 waves.
// Wave w: one q/k tile (w<2: q nt=w ; else k nt=w-2), then 4 v tiles
// (nt = w*4+i). All fragment strides are compile-time -> immediate offsets.
// ===========================================================================
__global__ void __launch_bounds__(128)
qkv_proj_kernel(const _Float16* __restrict__ xA,
                const _Float16* __restrict__ wqB, const _Float16* __restrict__ wkB,
                const _Float16* __restrict__ wvB,
                const float* __restrict__ bq, const float* __restrict__ bk,
                const float* __restrict__ bv,
                _Float16* __restrict__ qA, _Float16* __restrict__ kBt,
                _Float16* __restrict__ vB) {
  const int t = blockIdx.x;                      // global 16-row tile
  const int wave = threadIdx.x >> 5;
  const int lane = threadIdx.x & 31;
  const int n = lane & 15;
  const int mb = (lane < 16) ? 0 : 8;

  v16h a[8];
#pragma unroll
  for (int kk = 0; kk < 8; ++kk)
    a[kk] = *(const v16h*)(xA + ((size_t)t * 8 + kk) * 512 + lane * 16);

  // ---- section 1: q (waves 0,1) or k (waves 2,3); NT = 2 ----
  {
    const bool isQ = (wave < 2);
    const int nt = wave & 1;
    const _Float16* base = (isQ ? wqB : wkB) + (size_t)nt * 512 + lane * 16;
    const float bias = (isQ ? bq : bk)[nt * 16 + n];
    v8f acc = {};
#pragma unroll
    for (int kk = 0; kk < 8; ++kk)
      acc = WMMA_F16(a[kk], *(const v16h*)(base + kk * 2 * 512), acc);

    if (isQ) {                                    // q -> A-packed, one frag/tile
      const int laneA = ((n & 8) ? 16 : 0);
      const int j = (n & 7) + (nt ? 8 : 0);
#pragma unroll
      for (int v = 0; v < 8; ++v)
        qA[(size_t)t * 512 + (mb + v + laneA) * 16 + j] = (_Float16)(acc[v] + bias);
    } else {                                      // k -> B^T-packed per 16-key tile
      const int lhi = (n & 8) ? 16 : 0;
#pragma unroll
      for (int v = 0; v < 8; ++v)
        kBt[(size_t)t * 512 + (mb + v + lhi) * 16 + (n & 7) + (nt ? 8 : 0)] = (_Float16)(acc[v] + bias);
    }
  }

  // ---- section 2: v tiles, NT = 16 ----
  {
    const int bidx = t >> 8;                      // batch
    const int kt32 = (t & 255) >> 1;              // 32-key chunk in batch
    const int laneB = n + mb * 2;                 // key bit3 half
#pragma unroll
    for (int i = 0; i < 4; ++i) {
      const int nt = wave * 4 + i;
      const _Float16* base = wvB + (size_t)nt * 512 + lane * 16;
      v8f acc = {};
#pragma unroll
      for (int kk = 0; kk < 8; ++kk)
        acc = WMMA_F16(a[kk], *(const v16h*)(base + kk * 16 * 512), acc);
      const float bias = bv[nt * 16 + n];
      _Float16* dst = vB + (((size_t)bidx * 128 + kt32) * 16 + nt) * 512 + laneB * 16;
#pragma unroll
      for (int v = 0; v < 8; ++v)
        dst[((t & 1) ? 8 : 0) + v] = (_Float16)(acc[v] + bias);
    }
  }
}

// ===========================================================================
// Flash attention. Grid = B * (N/64), block = 256 (8 waves).
// Wave pair owns a 16-query strip; each wave accumulates 8 of 16 col tiles.
// K/V fragments are contiguous 32 B/lane loads from L2; only the C->A
// transpose of P uses a per-wave LDS buffer.
// ===========================================================================
__global__ void __launch_bounds__(256)
flash_attn_kernel(const _Float16* __restrict__ qA, const _Float16* __restrict__ kB,
                  const _Float16* __restrict__ vB, _Float16* __restrict__ attA) {
  const int b  = blockIdx.x >> 6;
  const int qt = blockIdx.x & 63;
  const int wave = threadIdx.x >> 5;
  const int lane = threadIdx.x & 31;
  const int strip = wave >> 1;
  const int colHalf = wave & 1;
  const int q0 = qt * 64 + strip * 16;
  const int qTile = (b * kN + q0) >> 4;           // global 16-row tile index

  __shared__ _Float16 sP[8][512];                 // per-wave prob tile (A-packed)

  const v16h aQ = *(const v16h*)(qA + (size_t)qTile * 512 + lane * 16);
  const _Float16* kStep = kB + (size_t)b * (kN / 16) * 512 + lane * 16;
  const _Float16* vStep = vB + ((size_t)b * (kN / 32) * 16 + colHalf * 8) * 512 + lane * 16;
  const float scale = 0.1767766952966369f;        // 1/sqrt(32)

  float mrow[8], lrow[8];
  v8f oacc[8];
#pragma unroll
  for (int v = 0; v < 8; ++v) { mrow[v] = -1e30f; lrow[v] = 0.f; oacc[v] = (v8f){}; }

  const int n = lane & 15;
  const int mb = (lane < 16) ? 0 : 8;
  const int laneAOff = (n & 8) ? 16 : 0;
  const int j0 = n & 7, j1 = (n & 7) + 8;
  _Float16* pw = sP[wave];

  for (int kt = 0; kt < kN / 32; ++kt, kStep += 1024, vStep += 16 * 512) {
    // prefetch next step's K/V (global_prefetch_b8)
    __builtin_prefetch(kStep + 1024, 0, 0);
    __builtin_prefetch(vStep + 16 * 512, 0, 0);

    // ---- S = Q K^T (K dim = D = 32: single WMMA per 16-key tile) ----
    v8f s0 = {}, s1 = {};
    s0 = WMMA_F16(aQ, *(const v16h*)(kStep), s0);
    s1 = WMMA_F16(aQ, *(const v16h*)(kStep + 512), s1);

    // ---- online softmax over this 32-key slab ----
    float alpha[8];
#pragma unroll
    for (int v = 0; v < 8; ++v) {
      float a0 = s0[v] * scale, a1 = s1[v] * scale;
      float mx = fmaxf(a0, a1);
#pragma unroll
      for (int msk = 1; msk < 16; msk <<= 1) mx = fmaxf(mx, __shfl_xor(mx, msk, 32));
      float mnew = fmaxf(mrow[v], mx);
      alpha[v] = __expf(mrow[v] - mnew);
      mrow[v] = mnew;
      float p0 = __expf(a0 - mnew), p1 = __expf(a1 - mnew);
      s0[v] = p0; s1[v] = p1;
      float rs = p0 + p1;
#pragma unroll
      for (int msk = 1; msk < 16; msk <<= 1) rs += __shfl_xor(rs, msk, 32);
      lrow[v] = lrow[v] * alpha[v] + rs;
    }
#pragma unroll
    for (int ct = 0; ct < 8; ++ct)
#pragma unroll
      for (int v = 0; v < 8; ++v) oacc[ct][v] *= alpha[v];

    // ---- C-layout f32 probs -> A-packed f16 via per-wave LDS ----
#pragma unroll
    for (int v = 0; v < 8; ++v) {
      int la = (mb + v) + laneAOff;               // key-col bit3 == n&8
      pw[la * 16 + j0] = (_Float16)s0[v];         // keys 0..15
      pw[la * 16 + j1] = (_Float16)s1[v];         // keys 16..31
    }
    v16h aP = *(const v16h*)(pw + lane * 16);

    // ---- O += P V (8 col tiles per wave, constant-offset fragment loads) ----
#pragma unroll
    for (int ct = 0; ct < 8; ++ct)
      oacc[ct] = WMMA_F16(aP, *(const v16h*)(vStep + ct * 512), oacc[ct]);
  }

  // ---- normalize, store attended in A-packed order for out_proj ----
  float rinv[8];
#pragma unroll
  for (int v = 0; v < 8; ++v) rinv[v] = 1.0f / lrow[v];
#pragma unroll
  for (int ct = 0; ct < 8; ++ct) {
    const int cidx = colHalf * 8 + ct;            // col tile 0..15
    const int kk = cidx >> 1;                     // 32-col K chunk
    const int jo = (cidx & 1) ? 8 : 0;            // kcol>=16 half
#pragma unroll
    for (int v = 0; v < 8; ++v) {
      float o = oacc[ct][v] * rinv[v];
      int laneA = (mb + v) + laneAOff;
      attA[((size_t)qTile * 8 + kk) * 512 + laneA * 16 + (n & 7) + jo] = (_Float16)o;
    }
  }
}

// ===========================================================================
// Output projection + bias + residual: out = x + att @ wo + bo (f32 out)
// ===========================================================================
__global__ void __launch_bounds__(128)
out_proj_kernel(const _Float16* __restrict__ attA, const _Float16* __restrict__ woB,
                const float* __restrict__ bo, const float* __restrict__ x,
                float* __restrict__ out) {
  const int t = blockIdx.x;                       // 16-row tile
  const int wave = threadIdx.x >> 5;
  const int lane = threadIdx.x & 31;
  const int n = lane & 15;
  const int mb = (lane < 16) ? 0 : 8;

  v16h a[8];
#pragma unroll
  for (int kk = 0; kk < 8; ++kk)
    a[kk] = *(const v16h*)(attA + ((size_t)t * 8 + kk) * 512 + lane * 16);

#pragma unroll
  for (int i = 0; i < 4; ++i) {
    const int ct = wave + i * 4;
    const _Float16* base = woB + (size_t)ct * 512 + lane * 16;
    v8f acc = {};
#pragma unroll
    for (int kk = 0; kk < 8; ++kk)
      acc = WMMA_F16(a[kk], *(const v16h*)(base + kk * 16 * 512), acc);
    const float bo_ = bo[ct * 16 + n];
#pragma unroll
    for (int v = 0; v < 8; ++v) {
      size_t idx = (size_t)(t * 16 + mb + v) * kC + ct * 16 + n;
      out[idx] = x[idx] + acc[v] + bo_;
    }
  }
}

// ===========================================================================
extern "C" void kernel_launch(void* const* d_in, const int* in_sizes, int n_in,
                              void* d_out, int out_size, void* d_ws, size_t ws_size,
                              hipStream_t stream) {
  (void)in_sizes; (void)n_in; (void)out_size; (void)ws_size;
  const float* x  = (const float*)d_in[0];
  const float* wq = (const float*)d_in[1];
  const float* bq = (const float*)d_in[2];
  const float* wk = (const float*)d_in[3];
  const float* bk = (const float*)d_in[4];
  const float* wv = (const float*)d_in[5];
  const float* bv = (const float*)d_in[6];
  const float* wo = (const float*)d_in[7];
  const float* bo = (const float*)d_in[8];
  float* out = (float*)d_out;

  char* ws = (char*)d_ws;
  _Float16* qA   = (_Float16*)(ws);                    // 1 MB
  _Float16* kBt  = (_Float16*)(ws + (1u  << 20));      // 1 MB
  _Float16* vB   = (_Float16*)(ws + (2u  << 20));      // 8 MB
  _Float16* attA = (_Float16*)(ws + (10u << 20));      // 8 MB
  _Float16* xA   = (_Float16*)(ws + (18u << 20));      // 8 MB
  _Float16* wqB  = (_Float16*)(ws + (26u << 20));      // 16 KB
  _Float16* wkB  = wqB + kC * kD;
  _Float16* wvB  = wkB + kC * kD;
  _Float16* woB  = wvB + kC * kC;

  pack_w_b_kernel<<<(kC * kD + 255) / 256, 256, 0, stream>>>(wq, wqB, kD);
  pack_w_b_kernel<<<(kC * kD + 255) / 256, 256, 0, stream>>>(wk, wkB, kD);
  pack_w_b_kernel<<<(kC * kC + 255) / 256, 256, 0, stream>>>(wv, wvB, kC);
  pack_w_b_kernel<<<(kC * kC + 255) / 256, 256, 0, stream>>>(wo, woB, kC);
  pack_x_a_kernel<<<(kRows * kC) / 256, 256, 0, stream>>>(x, xA);

  qkv_proj_kernel<<<kRows / 16, 128, 0, stream>>>(xA, wqB, wkB, wvB, bq, bk, bv,
                                                  qA, kBt, vB);
  flash_attn_kernel<<<kB * (kN / 64), 256, 0, stream>>>(qA, kBt, vB, attA);
  out_proj_kernel<<<kRows / 16, 128, 0, stream>>>(attA, woB, bo, x, out);
}
